// AdaConv2d_41755672052580
// MI455X (gfx1250) — matrix operand order repaired and checked by reading, not compiled, not run on hardware
//
#include <hip/hip_runtime.h>

typedef unsigned short u16;
typedef unsigned int   u32;
typedef unsigned long long u64;

typedef __attribute__((ext_vector_type(16))) __bf16 v16bf;
typedef __attribute__((ext_vector_type(8)))  float  v8f;
typedef __attribute__((ext_vector_type(4)))  int    v4i;

typedef __attribute__((address_space(1))) v4i* gv4i_p;   // global
typedef __attribute__((address_space(3))) v4i* lv4i_p;   // LDS

#define BB  32
#define CC  256
#define HH  64
#define WW  64
#define HWD 4096
#define OUTC 256
#define NCHUNK 72   // 9 taps * 8 ic-chunks of 32

__device__ __forceinline__ u16 f2bf(float f) {
    u32 u = __builtin_bit_cast(u32, f);
    u32 r = u + 0x7FFFu + ((u >> 16) & 1u);   // round-to-nearest-even
    return (u16)(r >> 16);
}
__device__ __forceinline__ float bf2f(u16 h) {
    u32 u = ((u32)h) << 16;
    return __builtin_bit_cast(float, u);
}
__device__ __forceinline__ int reflect64(int v) {
    return v < 0 ? -v : (v > 63 ? 126 - v : v);
}

// --- CDNA5 async global->LDS copy (ASYNCcnt path), 16B per lane ------------
// builtin signature (probe-derived): (v4i AS1* src, v4i AS3* dst, Ii, Ii)
// address-space casts must go through an integer (clang forbids direct
// C-style AS-changing pointer casts); low 32 bits of a flat LDS address
// are the LDS offset per the CDNA5 aperture rules.
__device__ __forceinline__ void async_copy16(const void* g, void* l) {
#if __has_builtin(__builtin_amdgcn_global_load_async_to_lds_b128)
    __builtin_amdgcn_global_load_async_to_lds_b128(
        (gv4i_p)(u64)g, (lv4i_p)(u32)(u64)l, 0, 0);
#else
    u32 lo = (u32)(u64)l;
    asm volatile("global_load_async_to_lds_b128 %0, %1, off"
                 :: "v"(lo), "v"((u64)g) : "memory");
#endif
}
__device__ __forceinline__ void wait_async0() {
#if __has_builtin(__builtin_amdgcn_s_wait_asynccnt)
    __builtin_amdgcn_s_wait_asynccnt(0);
#else
    asm volatile("s_wait_asynccnt 0" ::: "memory");
#endif
}

// ---------------------------------------------------------------------------
// Stage 0: conv_w [oc][ic][3][3] f32 -> bf16 hi/lo, layout [tap][oc][ic]
//          (ic contiguous so stage-2 A staging is contiguous 16B segments)
// ---------------------------------------------------------------------------
__global__ __launch_bounds__(256) void prep_w_kernel(
    const float* __restrict__ cw, u16* __restrict__ wHg, u16* __restrict__ wLg) {
    int i = blockIdx.x * 256 + threadIdx.x;       // exactly 9*256*256 threads
    int tap = i / (CC * OUTC);
    int rem = i - tap * (CC * OUTC);
    int oc  = rem >> 8;
    int ic  = rem & 255;
    float v = cw[(oc * CC + ic) * 9 + tap];
    u16 h = f2bf(v);
    wHg[i] = h;
    wLg[i] = f2bf(v - bf2f(h));
}

// ---------------------------------------------------------------------------
// Stage 1a: per-(b,c) instance-norm statistics  -> (mu, rsqrt(var+eps))
// ---------------------------------------------------------------------------
__global__ __launch_bounds__(256) void stats_kernel(
    const float* __restrict__ x, float2* __restrict__ stats) {
    __shared__ float r1[256];
    __shared__ float r2[256];
    const int bc  = blockIdx.x;
    const int tid = threadIdx.x;
    const float* xp = x + (size_t)bc * HWD;
    float s = 0.f, ss = 0.f;
#pragma unroll
    for (int j = 0; j < 4; ++j) {
        float4 v = ((const float4*)xp)[tid * 4 + j];
        s  += v.x + v.y + v.z + v.w;
        ss += v.x * v.x + v.y * v.y + v.z * v.z + v.w * v.w;
    }
    r1[tid] = s; r2[tid] = ss;
    __syncthreads();
    for (int o = 128; o > 0; o >>= 1) {
        if (tid < o) { r1[tid] += r1[tid + o]; r2[tid] += r2[tid + o]; }
        __syncthreads();
    }
    if (tid == 0) {
        float mu  = r1[0] * (1.f / HWD);
        float var = r2[0] * (1.f / HWD) - mu * mu;
        stats[bc] = make_float2(mu, rsqrtf(var + 1e-5f));
    }
}

// ---------------------------------------------------------------------------
// Stage 1b: dynamic depthwise 3x3 (reflect) + scale/bias, channel-last out.
//   Block: 32 channels x 8 rows x 64 w. lane = channel -> 64B coalesced
//   stores of y[b][hw][ic] bf16 hi/lo.
// ---------------------------------------------------------------------------
__global__ __launch_bounds__(256) void dw_kernel(
    const float* __restrict__ x, const float* __restrict__ wsp,
    const float* __restrict__ wpw, const float* __restrict__ bias,
    const float2* __restrict__ stats,
    u16* __restrict__ yTH, u16* __restrict__ yTL) {
    __shared__ float sm[32 * 641];               // 32 ch x (10 rows x 64) +pad

    const int tid = threadIdx.x;
    const int rt0 = blockIdx.x * 8;              // first output row of tile
    const int c0  = blockIdx.y * 32;
    const int b   = blockIdx.z;

    // fill halo tile: rows rt0-1 .. rt0+8 (reflected), 32 channels
#pragma unroll
    for (int j = 0; j < 80; ++j) {
        const int e   = j * 256 + tid;           // 0..20479
        const int cc  = e / 640;
        const int rem = e - cc * 640;
        const int r   = rem >> 6;
        const int w   = rem & 63;
        const int gr  = reflect64(rt0 + r - 1);
        sm[cc * 641 + rem] =
            x[((size_t)(b * CC + c0 + cc) * HH + gr) * WW + w];
    }
    __syncthreads();

    const int lane = tid & 31;                   // channel within chunk
    const int R    = tid >> 5;                   // row within tile (0..7)
    const int bc   = b * CC + c0 + lane;

    const float2 st = stats[bc];
    const float mu = st.x, rs = st.y;
    float wgt[9]; float wsum = 0.f;
#pragma unroll
    for (int t = 0; t < 9; ++t) { wgt[t] = wsp[bc * 9 + t]; wsum += wgt[t]; }
    const float wp = wpw[bc];
    const float bi = bias[bc];

    const float* chan = &sm[lane * 641];
    const size_t obase = ((size_t)b * HWD + (size_t)(rt0 + R) * WW) * CC + c0 + lane;

#pragma unroll 4
    for (int w = 0; w < 64; ++w) {
        float a = 0.f;
#pragma unroll
        for (int kh = 0; kh < 3; ++kh) {
            const float* rowp = chan + (R + kh) * 64;
#pragma unroll
            for (int kw = 0; kw < 3; ++kw)
                a += wgt[kh * 3 + kw] * rowp[reflect64(w + kw - 1)];
        }
        const float val = (a - mu * wsum) * rs * wp + bi;
        const u16 hb = f2bf(val);
        yTH[obase + (size_t)w * CC] = hb;
        yTL[obase + (size_t)w * CC] = f2bf(val - bf2f(hb));
    }
}

// ---------------------------------------------------------------------------
// Stage 2: dense 3x3 conv (reflect) as implicit GEMM, split-bf16 WMMA.
//   Block tile: M=128 oc, N=128 px (2 rows). 8 waves = 4(M) x 2(N).
//   Staging: global_load_async_to_lds_b128 (contiguous K-innermost layouts).
// ---------------------------------------------------------------------------
struct Frag { union { uint4 q[2]; v16bf v; }; };

__global__ __launch_bounds__(256) void conv_gemm_kernel(
    const u16* __restrict__ wTH, const u16* __restrict__ wTL,
    const u16* __restrict__ yTH, const u16* __restrict__ yTL,
    const float* __restrict__ cb, float* __restrict__ out) {
    __shared__ __attribute__((aligned(16))) u16 Ah[2][128 * 32];
    __shared__ __attribute__((aligned(16))) u16 Al[2][128 * 32];
    __shared__ __attribute__((aligned(16))) u16 Bh[2][128 * 32];
    __shared__ __attribute__((aligned(16))) u16 Bl[2][128 * 32];

    const int tid    = threadIdx.x;
    const int b      = blockIdx.y;
    const int h0     = blockIdx.x * 2;
    const int ocBase = blockIdx.z * 128;
    const int wave   = tid >> 5;
    const int lane   = tid & 31;
    const int wm     = wave >> 1;    // 0..3 -> 32 oc each
    const int wn     = wave & 1;     // 0..1 -> 64 px each (one image row)
    const int laneM  = lane & 15;
    const int hiH    = lane >> 4;

    v8f acc[2][4] = {};

    // issue one chunk's staging: 8 x 16B async copies per lane
    auto issueChunk = [&](int ck, int buf) {
        const int tap = ck >> 3;
        const int ic0 = (ck & 7) * 32;
        const int kh  = tap / 3, kw = tap - 3 * kh;
        // A: weights [tap][oc][ic]; 128 oc x 4 segs of 8 u16
#pragma unroll
        for (int j = 0; j < 2; ++j) {
            const int e   = j * 256 + tid;       // 0..511
            const int oc  = e >> 2, seg = e & 3;
            const size_t gb = (size_t)((tap * OUTC + ocBase + oc) * CC + ic0) + seg * 8;
            const int    lb = oc * 32 + seg * 8;
            async_copy16(wTH + gb, &Ah[buf][lb]);
            async_copy16(wTL + gb, &Al[buf][lb]);
        }
        // B: y channel-last [b][hw][ic]; 128 px x 4 segs of 8 u16
#pragma unroll
        for (int j = 0; j < 2; ++j) {
            const int e   = j * 256 + tid;       // 0..511
            const int px  = e >> 2, seg = e & 3;
            const int hh  = reflect64(h0 + (px >> 6) + kh - 1);
            const int ww  = reflect64((px & 63) + kw - 1);
            const size_t gb = (size_t)((b * HWD + hh * WW + ww) * CC + ic0) + seg * 8;
            const int    lb = px * 32 + seg * 8;
            async_copy16(yTH + gb, &Bh[buf][lb]);
            async_copy16(yTL + gb, &Bl[buf][lb]);
        }
    };

    auto computeChunk = [&](int buf) {
        Frag fah[2], fal[2];
#pragma unroll
        for (int mt = 0; mt < 2; ++mt) {
            const int row = (wm * 32 + mt * 16 + laneM) * 32 + 8 * hiH;
            const u16* pa = &Ah[buf][row];
            fah[mt].q[0] = *(const uint4*)pa;
            fah[mt].q[1] = *(const uint4*)(pa + 16);
            const u16* pl = &Al[buf][row];
            fal[mt].q[0] = *(const uint4*)pl;
            fal[mt].q[1] = *(const uint4*)(pl + 16);
        }
#pragma unroll
        for (int nt = 0; nt < 4; ++nt) {
            const int col = (wn * 64 + nt * 16 + laneM) * 32 + 8 * hiH;
            Frag fbh, fbl;
            const u16* pb = &Bh[buf][col];
            fbh.q[0] = *(const uint4*)pb;
            fbh.q[1] = *(const uint4*)(pb + 16);
            const u16* pq = &Bl[buf][col];
            fbl.q[0] = *(const uint4*)pq;
            fbl.q[1] = *(const uint4*)(pq + 16);
#pragma unroll
            for (int mt = 0; mt < 2; ++mt) {
                acc[mt][nt] = __builtin_amdgcn_wmma_f32_16x16x32_bf16(
                    false, fah[mt].v, false, fbh.v, (short)0, acc[mt][nt], false, false);
                acc[mt][nt] = __builtin_amdgcn_wmma_f32_16x16x32_bf16(
                    false, fah[mt].v, false, fbl.v, (short)0, acc[mt][nt], false, false);
                acc[mt][nt] = __builtin_amdgcn_wmma_f32_16x16x32_bf16(
                    false, fal[mt].v, false, fbh.v, (short)0, acc[mt][nt], false, false);
            }
        }
    };

    issueChunk(0, 0);
    wait_async0();
    __syncthreads();

#pragma unroll 1
    for (int ck = 0; ck < NCHUNK; ++ck) {
        if (ck + 1 < NCHUNK) issueChunk(ck + 1, (ck + 1) & 1);
        computeChunk(ck & 1);
        wait_async0();           // my async copies into buf^1 have landed
        __syncthreads();         // everyone's copies visible, reads of buf done
    }

    // epilogue: D layout -> lanes 0-15: M=r, N=lane ; lanes 16-31: M=r+8
#pragma unroll
    for (int mt = 0; mt < 2; ++mt)
#pragma unroll
        for (int nt = 0; nt < 4; ++nt) {
            const int px = wn * 64 + nt * 16 + laneM;
            const int h  = h0 + (px >> 6);
            const int w  = px & 63;
#pragma unroll
            for (int r = 0; r < 8; ++r) {
                const int oc = ocBase + wm * 32 + mt * 16 + 8 * hiH + r;
                out[((b * OUTC + oc) * HH + h) * WW + w] = acc[mt][nt][r] + cb[oc];
            }
        }
}

// ---------------------------------------------------------------------------
extern "C" void kernel_launch(void* const* d_in, const int* in_sizes, int n_in,
                              void* d_out, int out_size, void* d_ws, size_t ws_size,
                              hipStream_t stream) {
    const float* x    = (const float*)d_in[0];
    const float* wsp  = (const float*)d_in[1];
    const float* wpw  = (const float*)d_in[2];
    const float* bias = (const float*)d_in[3];
    const float* cw   = (const float*)d_in[4];
    const float* cb   = (const float*)d_in[5];
    float* out        = (float*)d_out;

    // workspace layout (~137 MB):
    //   yTH : [b][hw][ic] bf16 = 64 MB      yTL : 64 MB
    //   wTH : [tap][oc][ic] bf16 = 1.125 MB wTL : 1.125 MB
    //   stats: 8192 x float2 = 64 KB
    char* ws = (char*)d_ws;
    u16* yTH = (u16*)ws;
    u16* yTL = (u16*)(ws + 67108864ull);
    u16* wTH = (u16*)(ws + 134217728ull);
    u16* wTL = (u16*)(ws + 135397376ull);
    float2* stats = (float2*)(ws + 136577024ull);

    prep_w_kernel<<<2304, 256, 0, stream>>>(cw, wTH, wTL);
    stats_kernel<<<BB * CC, 256, 0, stream>>>(x, stats);
    dw_kernel<<<dim3(8, 8, BB), 256, 0, stream>>>(x, wsp, wpw, bias, stats, yTH, yTL);
    conv_gemm_kernel<<<dim3(HH / 2, BB, 2), 256, 0, stream>>>(wTH, wTL, yTH, yTL, cb, out);
}
